// ImprovedSpectralBlock_69569880260659
// MI455X (gfx1250) — matrix operand order repaired and checked by reading, not compiled, not run on hardware
//
#include <hip/hip_runtime.h>

// Problem sizes (fixed by the reference)
#define BB   4
#define TT   2048
#define DD   1024
#define NFQ  512
#define NBQ  4
#define MROWS (BB * TT)      // 8192
#define NFFT 4096
#define LOG2FFT 12
#define FREQ_SCALE 0.1f

typedef __bf16 bf16;
typedef __attribute__((ext_vector_type(16))) __bf16 v16bf;
typedef __attribute__((ext_vector_type(8)))  __bf16 v8bf;
typedef __attribute__((ext_vector_type(8)))  float  v8f;

__device__ __forceinline__ float sigm(float x) { return 1.0f / (1.0f + __expf(-x)); }

__device__ __forceinline__ v8bf ld8(const bf16* p) { return *(const v8bf*)p; }

__device__ __forceinline__ v16bf cat16(v8bf lo, v8bf hi) {
  return __builtin_shufflevector(lo, hi, 0,1,2,3,4,5,6,7,8,9,10,11,12,13,14,15);
}

__device__ __forceinline__ v8f wmma_bf16(v16bf a, v16bf b, v8f c) {
  // D = A(16x32) * B(32x16) + C, f32 accumulate
  return __builtin_amdgcn_wmma_f32_16x16x32_bf16(false, a, false, b, (short)0, c, false, false);
}

// A fragment: rows [aRow, aRow+16), K window [kBase, kBase+32). Row-major, leading dim lda.
// Per ISA 7.12.2: lane m=L&15, half=L>>4; slots 0..7 -> K=half*8+j, slots 8..15 -> K=16+half*8+j
__device__ __forceinline__ v16bf loadA(const bf16* A, int lda, int aRow, int kBase, int lane) {
  int m = lane & 15, half = lane >> 4;
  const bf16* p = A + (size_t)(aRow + m) * lda + kBase + half * 8;
  return cat16(ld8(p), ld8(p + 16));
}

// Epilogue modes: 0=bias, 1=sigmoid(bias+acc), 2=gelu_exact(bias+acc), 3=acc*scale, 4=acc+bias+resid
template <int EPI, bool OUTB>
__device__ __forceinline__ void storeTile(v8f acc, int mTop, int n, int N,
                                          const float* bias, float* outF, bf16* outB,
                                          float sc, const float* resid) {
  float bv = bias ? bias[n] : 0.0f;
#pragma unroll
  for (int r = 0; r < 8; ++r) {
    size_t idx = (size_t)(mTop + r) * N + n;
    float v = acc[r] + bv;
    if (EPI == 1) v = sigm(v);
    else if (EPI == 2) v = 0.5f * v * (1.0f + erff(v * 0.70710678118f));
    else if (EPI == 3) v = v * sc;
    else if (EPI == 4) v = v + resid[idx];
    if (OUTB) outB[idx] = (bf16)v;
    else      outF[idx] = v;
  }
}

// Wave-tiled GEMM: out[M,N] = epi(A[M,K] @ W[N,K]^T).
// Block = 8 waves = 128x128 tile; each wave owns 32x64 (8 WMMAs per K=32 step).
// The 128-col x 32-K weight tile (8KB) is double-buffered in LDS and filled with
// GLOBAL_LOAD_ASYNC_TO_LDS_B128 (ASYNCcnt path) overlapping the WMMAs of the
// previous K-step; fragments are consumed via ds_load_b128. A fragments come
// straight from global (overlapped with the async fill) with global_prefetch_b8.
#define BTILE_STRIDE 40   // 32 K elems + 8 pad to spread LDS banks
template <int EPI, bool OUTB>
__global__ void __launch_bounds__(256) gemm_bf16_k(const bf16* __restrict__ A,
                                                   const bf16* __restrict__ W,
                                                   const float* __restrict__ bias,
                                                   float* __restrict__ outF,
                                                   bf16* __restrict__ outB,
                                                   int M, int N, int K,
                                                   const float* __restrict__ scale,
                                                   const float* __restrict__ resid) {
  __shared__ bf16 tileB[2][128 * BTILE_STRIDE];   // 2 x 10KB (8KB data + pad) = 16KB data, 20KB total
  int tid = threadIdx.x;
  int lane = tid & 31;
  int wave = tid >> 5;
  int wm = wave & 3;      // 4 waves along M (4*32 = 128)
  int wn = wave >> 2;     // 2 waves along N (2*64 = 128)
  int nBlock = blockIdx.x * 128;
  int mBase = blockIdx.y * 128 + wm * 32;
  int nL = lane & 15, half = lane >> 4;

  // This thread's two 16B fill chunks: rows r0 and r0+64, same 16B part.
  int r0 = tid >> 2, p8 = (tid & 3) * 8;

  v8f acc[2][4];
#pragma unroll
  for (int i = 0; i < 2; ++i)
#pragma unroll
    for (int j = 0; j < 4; ++j) acc[i][j] = (v8f){};

  // Async fill of one K-step weight tile into tileB[buf] (unconditional, 2 issues/thread)
  auto fillB = [&](int buf, int k0) {
    unsigned l0 = (unsigned)(uintptr_t)&tileB[buf][r0 * BTILE_STRIDE + p8];
    const bf16* g0 = W + (size_t)(nBlock + r0) * K + k0 + p8;
    asm volatile("global_load_async_to_lds_b128 %0, %1, off"
                 :: "v"(l0), "v"((unsigned long long)(uintptr_t)g0) : "memory");
    unsigned l1 = (unsigned)(uintptr_t)&tileB[buf][(r0 + 64) * BTILE_STRIDE + p8];
    const bf16* g1 = W + (size_t)(nBlock + r0 + 64) * K + k0 + p8;
    asm volatile("global_load_async_to_lds_b128 %0, %1, off"
                 :: "v"(l1), "v"((unsigned long long)(uintptr_t)g1) : "memory");
  };

  fillB(0, 0);   // prologue

  int buf = 0;
  for (int k0 = 0; k0 < K; k0 += 32, buf ^= 1) {
    // A fragments + prefetch issued first so they overlap the async LDS fill
    v16bf a0 = loadA(A, K, mBase,      k0, lane);
    v16bf a1 = loadA(A, K, mBase + 16, k0, lane);
    if (k0 + 32 < K)
      __builtin_prefetch(A + (size_t)(mBase + nL) * K + k0 + 32, 0, 1);
    // my fill of this K-step is the only outstanding async op here
    asm volatile("s_wait_asynccnt 0" ::: "memory");
    __syncthreads();   // all fills visible; all waves done reading tileB[buf^1]
    if (k0 + 32 < K)
      fillB(buf ^ 1, k0 + 32);   // overlaps the WMMAs below
#pragma unroll
    for (int j = 0; j < 4; ++j) {
      const bf16* p = &tileB[buf][(wn * 64 + j * 16 + nL) * BTILE_STRIDE + half * 8];
      v16bf bfrag = cat16(*(const v8bf*)p, *(const v8bf*)(p + 16));
      acc[0][j] = wmma_bf16(a0, bfrag, acc[0][j]);
      acc[1][j] = wmma_bf16(a1, bfrag, acc[1][j]);
    }
  }

  float sc = (EPI == 3 && scale) ? scale[0] : 1.0f;
  int mOff = half * 8;
#pragma unroll
  for (int i = 0; i < 2; ++i)
#pragma unroll
    for (int j = 0; j < 4; ++j)
      storeTile<EPI, OUTB>(acc[i][j], mBase + i * 16 + mOff,
                           nBlock + wn * 64 + j * 16 + nL, N,
                           bias, outF, outB, sc, resid);
}

// f32 -> bf16 conversion (weights)
__global__ void f32_to_bf16_k(const float* __restrict__ src, bf16* __restrict__ dst, size_t n) {
  size_t i = (size_t)blockIdx.x * blockDim.x + threadIdx.x;
  size_t stride = (size_t)gridDim.x * blockDim.x;
  for (; i < n; i += stride) dst[i] = (bf16)src[i];
}

// LayerNorm over D=1024, one block (256 thr) per row, bf16 output
__global__ void __launch_bounds__(256) layernorm_k(const float* __restrict__ x,
                                                   const float* __restrict__ g,
                                                   const float* __restrict__ b,
                                                   bf16* __restrict__ out) {
  int row = blockIdx.x, tid = threadIdx.x;
  const float* xr = x + (size_t)row * DD;
  float vals[4], lsum = 0.f, lsq = 0.f;
#pragma unroll
  for (int i = 0; i < 4; ++i) {
    float v = xr[tid + i * 256];
    vals[i] = v; lsum += v; lsq += v * v;
  }
  __shared__ float s1[256], s2[256];
  s1[tid] = lsum; s2[tid] = lsq; __syncthreads();
  for (int off = 128; off > 0; off >>= 1) {
    if (tid < off) { s1[tid] += s1[tid + off]; s2[tid] += s2[tid + off]; }
    __syncthreads();
  }
  float mean = s1[0] * (1.0f / DD);
  float var  = s2[0] * (1.0f / DD) - mean * mean;
  float rs = rsqrtf(var + 1e-5f);
#pragma unroll
  for (int i = 0; i < 4; ++i) {
    int c = tid + i * 256;
    out[(size_t)row * DD + c] = (bf16)((vals[i] - mean) * rs * g[c] + b[c]);
  }
}

// Tiny projections: tg (1 x D) and band (4 x D), one block per row; sigmoid applied.
__global__ void __launch_bounds__(256) tgband_k(const bf16* __restrict__ xn,
                                                const float* __restrict__ Wtg, const float* __restrict__ btg,
                                                const float* __restrict__ Wband, const float* __restrict__ bband,
                                                float* __restrict__ tg_sig, float* __restrict__ band_sig) {
  int row = blockIdx.x, tid = threadIdx.x;
  const bf16* xr = xn + (size_t)row * DD;
  float part[5] = {0.f, 0.f, 0.f, 0.f, 0.f};
  for (int c = tid; c < DD; c += 256) {
    float xv = (float)xr[c];
    part[0] += xv * Wtg[c];
#pragma unroll
    for (int j = 0; j < NBQ; ++j) part[1 + j] += xv * Wband[(size_t)j * DD + c];
  }
  __shared__ float s[5][256];
#pragma unroll
  for (int j = 0; j < 5; ++j) s[j][tid] = part[j];
  __syncthreads();
  for (int off = 128; off > 0; off >>= 1) {
    if (tid < off) {
#pragma unroll
      for (int j = 0; j < 5; ++j) s[j][tid] += s[j][tid + off];
    }
    __syncthreads();
  }
  if (tid == 0) tg_sig[row] = sigm(s[0][0] + btg[0]);
  if (tid < NBQ) band_sig[(size_t)row * NBQ + tid] = sigm(s[1 + tid][0] + bband[tid]);
}

// Per-frequency constants: decay[f] = sigmoid(log_decay)*softmax(fi)*NF ; omega[f]
__global__ void __launch_bounds__(256) prep_freq_k(const float* __restrict__ fi,
                                                   const float* __restrict__ ldcy,
                                                   const float* __restrict__ freqs,
                                                   const float* __restrict__ dt,
                                                   float* __restrict__ decayArr,
                                                   float* __restrict__ omegaArr) {
  __shared__ float s[256];
  int tid = threadIdx.x;
  float p = 0.f;
  for (int i = tid; i < NFQ; i += 256) p += __expf(fi[i]);
  s[tid] = p; __syncthreads();
  for (int off = 128; off > 0; off >>= 1) {
    if (tid < off) s[tid] += s[tid + off];
    __syncthreads();
  }
  float denom = s[0];
  for (int i = tid; i < NFQ; i += 256) {
    decayArr[i] = sigm(ldcy[i]) * (__expf(fi[i]) / denom * (float)NFQ);
    omegaArr[i] = freqs[i] * FREQ_SCALE * (2.0f * sigm(dt[i]));
  }
}

// In-place radix-2 DIT FFT on LDS arrays (NFFT=4096, 256 threads). sign=-1 fwd, +1 inv.
__device__ void fft_inplace(float* re, float* im, float sign) {
  const int NT = 256;
  int tid = threadIdx.x;
  __syncthreads();
  for (int i = tid; i < NFFT; i += NT) {
    int j = (int)(__builtin_bitreverse32((unsigned)i) >> (32 - LOG2FFT));
    if (j > i) {
      float tr = re[i]; re[i] = re[j]; re[j] = tr;
      float ti = im[i]; im[i] = im[j]; im[j] = ti;
    }
  }
  __syncthreads();
  for (int len = 2; len <= NFFT; len <<= 1) {
    int half = len >> 1;
    float base = sign * 6.283185307179586f / (float)len;
    for (int idx = tid; idx < NFFT / 2; idx += NT) {
      int blk = idx / half, pos = idx - blk * half;
      int i0 = blk * len + pos, i1 = i0 + half;
      float ang = base * (float)pos;
      float wr = __cosf(ang), wi = __sinf(ang);
      float xr = re[i1], xi = im[i1];
      float tr = wr * xr - wi * xi, ti = wr * xi + wi * xr;
      float ur = re[i0], ui = im[i0];
      re[i0] = ur + tr; im[i0] = ui + ti;
      re[i1] = ur - tr; im[i1] = ui - ti;
    }
    __syncthreads();
  }
}

// Fused spectral conv: one block per (freq f, batch b).
// kernel[t] = exp(t*ln(decay) + prefix(log bandscale)) * ts[t] * e^{i(t+1)w}
// y = ifft(fft(u) * fft(kernel)) truncated to T; write [re|im] as bf16.
__global__ void __launch_bounds__(256) fft_conv_k(const float* __restrict__ sp,
                                                  const float* __restrict__ fg,
                                                  const float* __restrict__ tg_sig,
                                                  const float* __restrict__ band_sig,
                                                  const float* __restrict__ decayArr,
                                                  const float* __restrict__ omegaArr,
                                                  bf16* __restrict__ spec16) {
  extern __shared__ float smem[];
  float* Ar = smem;
  float* Ai = smem + NFFT;
  float* Br = smem + 2 * NFFT;
  float* Bi = smem + 3 * NFFT;
  const int NT = 256;
  int f = blockIdx.x, b = blockIdx.y, tid = threadIdx.x;
  int bandIdx = f / (NFQ / NBQ);
  float lndec = logf(decayArr[f]);
  float om = omegaArr[f];

  // 1) log band-scale, inclusive scan over T (Hillis-Steele ping-pong in Ar/Ai)
  float myl[TT / 256];
  {
    int i = 0;
    for (int t = tid; t < TT; t += NT, ++i) {
      float bs = band_sig[((size_t)b * TT + t) * NBQ + bandIdx];
      myl[i] = logf(bs);
      Ar[t] = myl[i];
    }
  }
  __syncthreads();
  float* src = Ar; float* dst = Ai;
  for (int off = 1; off < TT; off <<= 1) {
    for (int t = tid; t < TT; t += NT)
      dst[t] = src[t] + (t >= off ? src[t - off] : 0.0f);
    __syncthreads();
    float* tmp = src; src = dst; dst = tmp;
  }
  // 2) build conv kernel into B (exclusive prefix = inclusive - own)
  {
    int i = 0;
    for (int t = tid; t < TT; t += NT, ++i) {
      float excl = src[t] - myl[i];
      float ts = tg_sig[(size_t)b * TT + t];
      float mag = __expf((float)t * lndec + excl) * ts;
      float ph = (float)(t + 1) * om;
      Br[t] = mag * __cosf(ph);
      Bi[t] = mag * __sinf(ph);
    }
    for (int t = TT + tid; t < NFFT; t += NT) { Br[t] = 0.f; Bi[t] = 0.f; }
  }
  __syncthreads();
  // 3) build u into A
  for (int t = tid; t < TT; t += NT) {
    size_t rowoff = ((size_t)b * TT + t);
    float fgv = fg[rowoff * NFQ + f];
    Ar[t] = sp[rowoff * 2 * NFQ + f]        * fgv;
    Ai[t] = sp[rowoff * 2 * NFQ + NFQ + f]  * fgv;
  }
  for (int t = TT + tid; t < NFFT; t += NT) { Ar[t] = 0.f; Ai[t] = 0.f; }

  // 4) FFTs, pointwise product, inverse FFT
  fft_inplace(Ar, Ai, -1.0f);
  fft_inplace(Br, Bi, -1.0f);
  for (int i = tid; i < NFFT; i += NT) {
    float ar = Ar[i], ai = Ai[i], br = Br[i], bi = Bi[i];
    Ar[i] = ar * br - ai * bi;
    Ai[i] = ar * bi + ai * br;
  }
  fft_inplace(Ar, Ai, 1.0f);

  // 5) write bf16 spectral_out = [y.real | y.imag]
  const float inv = 1.0f / (float)NFFT;
  for (int t = tid; t < TT; t += NT) {
    size_t rowoff = ((size_t)b * TT + t) * (2 * NFQ);
    spec16[rowoff + f]       = (bf16)(Ar[t] * inv);
    spec16[rowoff + NFQ + f] = (bf16)(Ai[t] * inv);
  }
}

// x2 = x + (yproj*gate + concept*cg_sig)/2 ; then LN2 -> xn2 (bf16). cg buffer becomes x2 in-place.
__global__ void __launch_bounds__(256) combine_ln2_k(const float* __restrict__ x,
                                                     const float* __restrict__ yproj,
                                                     const float* __restrict__ gate,
                                                     const float* __restrict__ concept,
                                                     float* __restrict__ cg_x2,
                                                     const float* __restrict__ g2,
                                                     const float* __restrict__ b2,
                                                     bf16* __restrict__ xn2) {
  int row = blockIdx.x, tid = threadIdx.x;
  size_t base = (size_t)row * DD;
  float vals[4], lsum = 0.f, lsq = 0.f;
#pragma unroll
  for (int i = 0; i < 4; ++i) {
    size_t idx = base + tid + i * 256;
    float cp = concept[idx] * cg_x2[idx];                 // concept * sigmoid(cg)
    float v = x[idx] + 0.5f * (yproj[idx] * gate[idx] + cp);
    vals[i] = v; lsum += v; lsq += v * v;
  }
  __shared__ float s1[256], s2[256];
  s1[tid] = lsum; s2[tid] = lsq; __syncthreads();
  for (int off = 128; off > 0; off >>= 1) {
    if (tid < off) { s1[tid] += s1[tid + off]; s2[tid] += s2[tid + off]; }
    __syncthreads();
  }
  float mean = s1[0] * (1.0f / DD);
  float var  = s2[0] * (1.0f / DD) - mean * mean;
  float rs = rsqrtf(var + 1e-5f);
#pragma unroll
  for (int i = 0; i < 4; ++i) {
    int c = tid + i * 256;
    size_t idx = base + c;
    cg_x2[idx] = vals[i];                                  // x2 (residual for final GEMM)
    xn2[idx] = (bf16)((vals[i] - mean) * rs * g2[c] + b2[c]);
  }
}

extern "C" void kernel_launch(void* const* d_in, const int* in_sizes, int n_in,
                              void* d_out, int out_size, void* d_ws, size_t ws_size,
                              hipStream_t stream) {
  (void)in_sizes; (void)n_in; (void)out_size; (void)ws_size;
  // Inputs (setup_inputs order, all float32)
  const float* x        = (const float*)d_in[0];
  const float* norm_g   = (const float*)d_in[1];
  const float* norm_b   = (const float*)d_in[2];
  const float* Wconcept = (const float*)d_in[3];
  const float* bconcept = (const float*)d_in[4];
  const float* Wcg      = (const float*)d_in[5];
  const float* bcg      = (const float*)d_in[6];
  const float* Wfg      = (const float*)d_in[7];
  const float* bfg      = (const float*)d_in[8];
  const float* Wtg      = (const float*)d_in[9];
  const float* btg      = (const float*)d_in[10];
  const float* Wband    = (const float*)d_in[11];
  const float* bband    = (const float*)d_in[12];
  const float* Wsp      = (const float*)d_in[13];
  const float* fimp     = (const float*)d_in[14];
  const float* log_dec  = (const float*)d_in[15];
  const float* freqs    = (const float*)d_in[16];
  const float* dt       = (const float*)d_in[17];
  const float* Wgate    = (const float*)d_in[18];
  const float* bgate    = (const float*)d_in[19];
  const float* Wfs      = (const float*)d_in[20];
  const float* out_sc   = (const float*)d_in[21];
  const float* norm2_g  = (const float*)d_in[22];
  const float* norm2_b  = (const float*)d_in[23];
  const float* W1       = (const float*)d_in[24];
  const float* b1       = (const float*)d_in[25];
  const float* W2       = (const float*)d_in[26];
  const float* b2       = (const float*)d_in[27];
  float* out = (float*)d_out;

  // Workspace carve (256B aligned)
  char* ws = (char*)d_ws;
  size_t off = 0;
  auto carve = [&](size_t bytes) -> void* {
    void* p = ws + off;
    off += (bytes + 255) & ~(size_t)255;
    return p;
  };
  const size_t MD  = (size_t)MROWS * DD;      // 8M elems
  const size_t MNF = (size_t)MROWS * NFQ;     // 4M elems
  bf16*  xn16    = (bf16*)carve(MD * 2);             // xn, later reused as xn2
  bf16*  spec16  = (bf16*)carve(MD * 2);             // (M, 2*NF) bf16
  float* fgate   = (float*)carve(MNF * 4);
  float* spF     = (float*)carve(MD * 4);            // sp f32, later reused as yproj
  float* bufA    = (float*)carve(MD * 4);            // concept ... then h[0:32MB]
  float* gateB   = (float*)carve(MD * 4);            // gate   ... then h[32:64MB]
  float* bufB    = (float*)carve(MD * 4);            // sigmoid(cg), then x2
  float* tg_sig  = (float*)carve((size_t)MROWS * 4);
  float* band_sg = (float*)carve((size_t)MROWS * NBQ * 4);
  float* decayA  = (float*)carve((size_t)NFQ * 4);
  float* omegaA  = (float*)carve((size_t)NFQ * 4);
  bf16* wC16  = (bf16*)carve((size_t)DD * DD * 2);
  bf16* wCg16 = (bf16*)carve((size_t)DD * DD * 2);
  bf16* wFg16 = (bf16*)carve((size_t)NFQ * DD * 2);
  bf16* wSp16 = (bf16*)carve((size_t)2 * NFQ * DD * 2);
  bf16* wGt16 = (bf16*)carve((size_t)DD * DD * 2);
  bf16* wFs16 = (bf16*)carve((size_t)DD * 2 * NFQ * 2);
  bf16* w116  = (bf16*)carve((size_t)4 * DD * DD * 2);
  bf16* w216  = (bf16*)carve((size_t)DD * 4 * DD * 2);
  bf16* hBuf  = (bf16*)bufA;                          // 64MB region = bufA + gateB

  // 0) weights -> bf16
  auto cvt = [&](const float* s, bf16* d, size_t n) {
    f32_to_bf16_k<<<2048, 256, 0, stream>>>(s, d, n);
  };
  cvt(Wconcept, wC16,  (size_t)DD * DD);
  cvt(Wcg,      wCg16, (size_t)DD * DD);
  cvt(Wfg,      wFg16, (size_t)NFQ * DD);
  cvt(Wsp,      wSp16, (size_t)2 * NFQ * DD);
  cvt(Wgate,    wGt16, (size_t)DD * DD);
  cvt(Wfs,      wFs16, (size_t)DD * 2 * NFQ);
  cvt(W1,       w116,  (size_t)4 * DD * DD);
  cvt(W2,       w216,  (size_t)DD * 4 * DD);

  // 1) LN1 -> xn (bf16)
  layernorm_k<<<MROWS, 256, 0, stream>>>(x, norm_g, norm_b, xn16);

  // 2) frequency constants
  prep_freq_k<<<1, 256, 0, stream>>>(fimp, log_dec, freqs, dt, decayA, omegaA);

  // 3) WMMA GEMMs from xn (block tile 128x128)
  dim3 gD(DD / 128, MROWS / 128);
  gemm_bf16_k<0,false><<<gD, 256, 0, stream>>>(xn16, wC16,  bconcept, bufA, nullptr, MROWS, DD, DD, nullptr, nullptr);
  gemm_bf16_k<1,false><<<gD, 256, 0, stream>>>(xn16, wCg16, bcg,      bufB, nullptr, MROWS, DD, DD, nullptr, nullptr);
  dim3 gNF(NFQ / 128, MROWS / 128);
  gemm_bf16_k<1,false><<<gNF, 256, 0, stream>>>(xn16, wFg16, bfg,     fgate, nullptr, MROWS, NFQ, DD, nullptr, nullptr);
  dim3 g2N(2 * NFQ / 128, MROWS / 128);
  gemm_bf16_k<0,false><<<g2N, 256, 0, stream>>>(xn16, wSp16, nullptr, spF,  nullptr, MROWS, 2 * NFQ, DD, nullptr, nullptr);
  gemm_bf16_k<1,false><<<gD, 256, 0, stream>>>(xn16, wGt16, bgate,    gateB, nullptr, MROWS, DD, DD, nullptr, nullptr);

  // 4) tiny projections (sigmoid applied)
  tgband_k<<<MROWS, 256, 0, stream>>>(xn16, Wtg, btg, Wband, bband, tg_sig, band_sg);

  // 5) spectral convolution via FFT (one block per (f,b)); 64KB dynamic LDS
  fft_conv_k<<<dim3(NFQ, BB), 256, 4 * NFFT * sizeof(float), stream>>>(
      spF, fgate, tg_sig, band_sg, decayA, omegaA, spec16);

  // 6) y_proj = (spectral @ Wfs^T) * output_scale   (reuses spF region)
  gemm_bf16_k<3,false><<<gD, 256, 0, stream>>>(spec16, wFs16, nullptr, spF, nullptr, MROWS, DD, 2 * NFQ, out_sc, nullptr);

  // 7) combine + LN2 (x2 written into bufB; xn2 reuses xn16)
  combine_ln2_k<<<MROWS, 256, 0, stream>>>(x, spF, gateB, bufA, bufB, norm2_g, norm2_b, xn16);

  // 8) MLP: h = gelu(xn2 @ W1^T + b1) (bf16 into reused 64MB region)
  dim3 g4D(4 * DD / 128, MROWS / 128);
  gemm_bf16_k<2,true><<<g4D, 256, 0, stream>>>(xn16, w116, b1, nullptr, hBuf, MROWS, 4 * DD, DD, nullptr, nullptr);

  // 9) out = x2 + h @ W2^T + b2
  gemm_bf16_k<4,false><<<gD, 256, 0, stream>>>(hBuf, w216, b2, out, nullptr, MROWS, DD, 4 * DD, nullptr, bufB);
}
